// GPT2Nucleus_61710090109593
// MI455X (gfx1250) — matrix operand order repaired and checked by reading, not compiled, not run on hardware
//
#include <hip/hip_runtime.h>
#include <hip/hip_bf16.h>
#include <math.h>

// GPT-2 forward for MI455X (gfx1250): bf16 WMMA GEMMs + flash attention.
// B=4, T=1024, E=1024, H=16, D=64, L=12, FF=4096.

#define BB  4
#define TT  1024
#define EE  1024
#define HH  16
#define DD  64
#define LLY 12
#define FFN 4096

typedef __attribute__((ext_vector_type(16))) __bf16 v16bf;
typedef __attribute__((ext_vector_type(8)))  float  v8f;

union Frag16 {
  v16bf v;
  unsigned short u[16];
  uint4 q[2];
};

// Pack two fp32 into packed bf16 (truncate) -- single v_perm_b32.
__device__ __forceinline__ unsigned int pk2bf(float lo, float hi) {
  return __builtin_amdgcn_perm(__float_as_uint(hi), __float_as_uint(lo),
                               0x07060302u);
}
__device__ __forceinline__ unsigned short f2bf(float f) {
  return (unsigned short)(__float_as_uint(f) >> 16);
}

// ---------------------------------------------------------------------------
// x[b,t,:] = tok_emb[tokens[b,t]] + pos_emb[0,t,:]
// ---------------------------------------------------------------------------
__global__ __launch_bounds__(256) void embed_kernel(
    const int* __restrict__ tokens, const float* __restrict__ tok_emb,
    const float* __restrict__ pos_emb, float* __restrict__ x) {
  const int row = blockIdx.x;            // b*T + t
  const int t   = row & (TT - 1);
  const int tok = tokens[row];
  const float4* te = (const float4*)(tok_emb + (size_t)tok * EE);
  const float4* pe = (const float4*)(pos_emb + (size_t)t * EE);
  float4* xr = (float4*)(x + (size_t)row * EE);
  const int i = threadIdx.x;             // 256 threads x float4 == 1024 floats
  const float4 a = te[i], b = pe[i];
  xr[i] = make_float4(a.x + b.x, a.y + b.y, a.z + b.z, a.w + b.w);
}

// ---------------------------------------------------------------------------
// LayerNorm over last dim (E=1024), one row per block (256 threads, 4/thread)
// ---------------------------------------------------------------------------
__global__ __launch_bounds__(256) void layernorm_kernel(
    const float* __restrict__ x, const float* __restrict__ g,
    const float* __restrict__ b, float* __restrict__ y) {
  const int row = blockIdx.x;
  const int i = threadIdx.x;
  const float4 v4 = ((const float4*)(x + (size_t)row * EE))[i];
  float s  = v4.x + v4.y + v4.z + v4.w;
  float s2 = v4.x * v4.x + v4.y * v4.y + v4.z * v4.z + v4.w * v4.w;
#pragma unroll
  for (int off = 16; off > 0; off >>= 1) {
    s  += __shfl_xor(s,  off);
    s2 += __shfl_xor(s2, off);
  }
  __shared__ float red[2][8];
  const int wave = i >> 5, lane = i & 31;
  if (lane == 0) { red[0][wave] = s; red[1][wave] = s2; }
  __syncthreads();
  s = 0.f; s2 = 0.f;
#pragma unroll
  for (int w = 0; w < 8; ++w) { s += red[0][w]; s2 += red[1][w]; }
  const float mean = s * (1.f / EE);
  const float var  = s2 * (1.f / EE) - mean * mean;
  const float rstd = rsqrtf(var + 1e-5f);
  const float4 g4 = ((const float4*)g)[i];
  const float4 b4 = ((const float4*)b)[i];
  float4 o;
  o.x = (v4.x - mean) * rstd * g4.x + b4.x;
  o.y = (v4.y - mean) * rstd * g4.y + b4.y;
  o.z = (v4.z - mean) * rstd * g4.z + b4.z;
  o.w = (v4.w - mean) * rstd * g4.w + b4.w;
  ((float4*)(y + (size_t)row * EE))[i] = o;
}

// ---------------------------------------------------------------------------
// fp32 -> bf16 bulk convert (8 elements/thread). SCALE folds in 1/sqrt(D).
// ---------------------------------------------------------------------------
template <bool SCALE>
__global__ __launch_bounds__(256) void convert_kernel(
    const float* __restrict__ src, unsigned short* __restrict__ dst) {
  const size_t i = (size_t)blockIdx.x * 256 + threadIdx.x;
  float4 a = ((const float4*)src)[2 * i];
  float4 b = ((const float4*)src)[2 * i + 1];
  if constexpr (SCALE) {
    a.x *= 0.125f; a.y *= 0.125f; a.z *= 0.125f; a.w *= 0.125f;
    b.x *= 0.125f; b.y *= 0.125f; b.z *= 0.125f; b.w *= 0.125f;
  }
  uint4 o;
  o.x = pk2bf(a.x, a.y); o.y = pk2bf(a.z, a.w);
  o.z = pk2bf(b.x, b.y); o.w = pk2bf(b.z, b.w);
  ((uint4*)dst)[i] = o;
}

// ---------------------------------------------------------------------------
// V [B,T,E] fp32 -> Vt [B,H,D,T] bf16 (per-head transpose via LDS tile).
// Grid: (B*H, T/64). Tile = 64 t x 64 d.
// ---------------------------------------------------------------------------
__global__ __launch_bounds__(256) void v_transpose_kernel(
    const float* __restrict__ v, unsigned short* __restrict__ vt) {
  const int b = blockIdx.x / HH, h = blockIdx.x % HH;
  const int t0 = blockIdx.y * 64;
  __shared__ unsigned short tile[64][72];   // [d][t] bf16 (+pad)
#pragma unroll
  for (int i = 0; i < 4; ++i) {
    const int lin = i * 256 + threadIdx.x;  // 1024 float4 tasks
    const int t = lin >> 4, d4 = (lin & 15) * 4;
    const float4 f = *(const float4*)(v + ((size_t)(b * TT + t0 + t)) * EE + h * DD + d4);
    tile[d4 + 0][t] = f2bf(f.x);
    tile[d4 + 1][t] = f2bf(f.y);
    tile[d4 + 2][t] = f2bf(f.z);
    tile[d4 + 3][t] = f2bf(f.w);
  }
  __syncthreads();
#pragma unroll
  for (int i = 0; i < 2; ++i) {
    const int lin = i * 256 + threadIdx.x;  // 512 uint4 tasks
    const int d = lin >> 3, t8 = (lin & 7) * 8;
    const uint4 o = *(const uint4*)&tile[d][t8];
    *(uint4*)(vt + ((size_t)((b * HH + h) * DD + d)) * TT + t0 + t8) = o;
  }
}

// ---------------------------------------------------------------------------
// C[M,N] = act(A[M,K] @ Bw[K,N] + bias) + residual
// 128x128 block tile, 8 waves, wave = 32x64 (2x4 WMMA tiles).
// Double-buffered LDS; A stored m-major/k-minor, B stored TRANSPOSED
// (n-major/k-minor) so every fragment is two 16B ds_load_b128.
// ACT: 0 = none, 1 = exact GELU.
// ---------------------------------------------------------------------------
template <int ACT, bool HAS_BIAS, bool HAS_RES>
__global__ __launch_bounds__(256) void gemm_kernel(
    const float* __restrict__ A, const float* __restrict__ Bw,
    const float* __restrict__ bias, const float* __restrict__ residual,
    float* __restrict__ C, int M, int N, int K) {
  __shared__ unsigned short As[2][128][40];  // [buf][m][k] bf16 (+pad)
  __shared__ unsigned short Bt[2][128][40];  // [buf][n][k] bf16 (+pad)

  const int tid  = threadIdx.x;
  const int lane = tid & 31;
  const int wave = tid >> 5;
  const int wrow = wave >> 1;                // 0..3 : 32-row slab
  const int wcol = wave & 1;                 // 0..1 : 64-col slab
  const int bm = blockIdx.y * 128;
  const int bn = blockIdx.x * 128;
  const int nlo = lane & 15, hi = lane >> 4;

  auto stage = [&](int buf, int k0) {
#pragma unroll
    for (int i = 0; i < 4; ++i) {
      const int lin = tid * 4 + i;
      const int r = lin >> 3, c = (lin & 7) * 4;
      const float4 f = *(const float4*)(A + (size_t)(bm + r) * K + (k0 + c));
      *(uint2*)&As[buf][r][c] = make_uint2(pk2bf(f.x, f.y), pk2bf(f.z, f.w));
    }
#pragma unroll
    for (int i = 0; i < 4; ++i) {
      const int lin = tid * 4 + i;
      const int n = lin & 127, kk = (lin >> 7) * 4;
      const float* bp = Bw + (size_t)(k0 + kk) * N + bn + n;
      const float b0 = bp[0];
      const float b1 = bp[(size_t)N];
      const float b2 = bp[(size_t)2 * N];
      const float b3 = bp[(size_t)3 * N];
      *(uint2*)&Bt[buf][n][kk] = make_uint2(pk2bf(b0, b1), pk2bf(b2, b3));
    }
  };

  v8f acc[2][4];
#pragma unroll
  for (int i = 0; i < 2; ++i)
#pragma unroll
    for (int j = 0; j < 4; ++j)
      acc[i][j] = (v8f){0.f, 0.f, 0.f, 0.f, 0.f, 0.f, 0.f, 0.f};

  stage(0, 0);
  int buf = 0;
  for (int k0 = 0; k0 < K; k0 += 32) {
    __syncthreads();
    if (k0 + 32 < K) stage(buf ^ 1, k0 + 32);
    if (k0 + 64 < K) {
      __builtin_prefetch(A + (size_t)(bm + (tid >> 1)) * K + k0 + 64, 0, 1);
      __builtin_prefetch(Bw + (size_t)(k0 + 64 + (tid >> 3)) * N + bn + (tid & 7) * 16, 0, 1);
    }

    Frag16 afr[2];
#pragma unroll
    for (int rt = 0; rt < 2; ++rt) {
      const int m  = wrow * 32 + rt * 16 + nlo;
      const int kb = hi * 8;
      afr[rt].q[0] = *(const uint4*)&As[buf][m][kb];
      afr[rt].q[1] = *(const uint4*)&As[buf][m][16 + kb];
    }
    Frag16 bfr[4];
#pragma unroll
    for (int nt = 0; nt < 4; ++nt) {
      const int n   = wcol * 64 + nt * 16 + nlo;
      const int kb2 = hi * 8;
      bfr[nt].q[0] = *(const uint4*)&Bt[buf][n][kb2];
      bfr[nt].q[1] = *(const uint4*)&Bt[buf][n][kb2 + 16];
    }
#pragma unroll
    for (int rt = 0; rt < 2; ++rt)
#pragma unroll
      for (int nt = 0; nt < 4; ++nt)
        acc[rt][nt] = __builtin_amdgcn_wmma_f32_16x16x32_bf16(
            false, afr[rt].v, false, bfr[nt].v, (short)0, acc[rt][nt],
            false, false);
    buf ^= 1;
  }

#pragma unroll
  for (int rt = 0; rt < 2; ++rt) {
#pragma unroll
    for (int nt = 0; nt < 4; ++nt) {
      const int gcol = bn + wcol * 64 + nt * 16 + nlo;
      float bv = 0.f;
      if constexpr (HAS_BIAS) bv = bias[gcol];
#pragma unroll
      for (int r = 0; r < 8; ++r) {
        const int grow = bm + wrow * 32 + rt * 16 + r + 8 * hi;
        float vv = acc[rt][nt][r] + bv;
        if constexpr (ACT == 1) vv = 0.5f * vv * (1.f + erff(vv * 0.70710678118f));
        const size_t idx = (size_t)grow * N + gcol;
        if constexpr (HAS_RES) vv += residual[idx];
        C[idx] = vv;
      }
    }
  }
}

// ---------------------------------------------------------------------------
// One 32-key flash-attention step for a 16-row query tile.
// MASK=false: interior block (no causal test). MASK=true: diagonal block.
// Row-sum of P comes from an extra WMMA against an all-ones B fragment
// (lands lane-locally in C layout -> no shuffle reduction needed).
// ---------------------------------------------------------------------------
template <bool MASK>
__device__ __forceinline__ void attn_step(
    int js, int qs, int nlo, int hi,
    const unsigned short* __restrict__ kbase,   // k + (b*T)*E + h*D
    const unsigned short* __restrict__ vbase,   // vt + ((b*H+h)*D)*T
    const Frag16 (&qf)[2], const Frag16& onesf,
    v8f (&o)[4], float (&mrow)[8], float (&lrow)[8],
    unsigned short (*__restrict__ ps)[40]) {    // this wave's [16][40] P tile
  v8f s[2];
  s[0] = (v8f){0.f, 0.f, 0.f, 0.f, 0.f, 0.f, 0.f, 0.f};
  s[1] = s[0];
  const int kb2 = hi * 8;
#pragma unroll
  for (int half = 0; half < 2; ++half) {
    const unsigned short* krow = kbase + ((size_t)(js + half * 16 + nlo)) * EE;
#pragma unroll
    for (int c = 0; c < 2; ++c) {
      Frag16 kf;
      kf.q[0] = *(const uint4*)(krow + c * 32 + kb2);
      kf.q[1] = *(const uint4*)(krow + c * 32 + kb2 + 16);
      s[half] = __builtin_amdgcn_wmma_f32_16x16x32_bf16(
          false, qf[c].v, false, kf.v, (short)0, s[half], false, false);
    }
  }
  float mx[8];
#pragma unroll
  for (int r = 0; r < 8; ++r) {
    if constexpr (MASK) {
      const int row = qs + r + 8 * hi;
      float s0 = s[0][r]; if (js + nlo > row)      s0 = -1e30f;
      float s1 = s[1][r]; if (js + 16 + nlo > row) s1 = -1e30f;
      s[0][r] = s0; s[1][r] = s1;
      mx[r] = fmaxf(s0, s1);
    } else {
      mx[r] = fmaxf(s[0][r], s[1][r]);
    }
  }
#pragma unroll
  for (int off = 1; off < 16; off <<= 1)
#pragma unroll
    for (int r = 0; r < 8; ++r)
      mx[r] = fmaxf(mx[r], __shfl_xor(mx[r], off));

  float al[8];
#pragma unroll
  for (int r = 0; r < 8; ++r) {
    const float mn = fmaxf(mrow[r], mx[r]);
    al[r] = __expf(mrow[r] - mn);
    mrow[r] = mn;
    ps[r + 8 * hi][nlo]      = f2bf(__expf(s[0][r] - mn));
    ps[r + 8 * hi][16 + nlo] = f2bf(__expf(s[1][r] - mn));
  }
#pragma unroll
  for (int nt = 0; nt < 4; ++nt)
#pragma unroll
    for (int r = 0; r < 8; ++r)
      o[nt][r] *= al[r];

  // Wave-private LDS round trip: drain DS counter before re-reading P
  asm volatile("s_wait_dscnt 0" ::: "memory");

  Frag16 pf;                        // P as 16x32 A-fragment (vector reads)
  {
    const int kb = hi * 8;
    pf.q[0] = *(const uint4*)&ps[nlo][kb];
    pf.q[1] = *(const uint4*)&ps[nlo][16 + kb];
  }
  // Row sums via P @ ones -- lane-local result, no cross-lane reduce.
  v8f rsv = __builtin_amdgcn_wmma_f32_16x16x32_bf16(
      false, pf.v, false, onesf.v, (short)0,
      (v8f){0.f, 0.f, 0.f, 0.f, 0.f, 0.f, 0.f, 0.f}, false, false);
#pragma unroll
  for (int r = 0; r < 8; ++r) lrow[r] = lrow[r] * al[r] + rsv[r];

  // O += P @ V : V-fragment from Vt[b,h,d,t] -> contiguous along t
#pragma unroll
  for (int nt = 0; nt < 4; ++nt) {
    Frag16 vf;
    const unsigned short* vrow = vbase + ((size_t)(nt * 16 + nlo)) * TT + js + kb2;
    vf.q[0] = *(const uint4*)(vrow);
    vf.q[1] = *(const uint4*)(vrow + 16);
    o[nt] = __builtin_amdgcn_wmma_f32_16x16x32_bf16(
        false, pf.v, false, vf.v, (short)0, o[nt], false, false);
  }
}

// ---------------------------------------------------------------------------
// Causal flash attention, all-bf16 operands.
//   q, k : bf16 [B,T,E]  (q pre-scaled by 1/sqrt(D))
//   vt   : bf16 [B,H,D,T] (pre-transposed)
// Grid: (B*H, T/128), 256 threads = 8 waves; each wave = 16 query rows.
// Interior key blocks run mask-free; exactly one diagonal block is masked.
// ---------------------------------------------------------------------------
__global__ __launch_bounds__(256) void attn_kernel(
    const unsigned short* __restrict__ q, const unsigned short* __restrict__ k,
    const unsigned short* __restrict__ vt, float* __restrict__ y) {
  const int b = blockIdx.x / HH;
  const int h = blockIdx.x % HH;
  const int wave = threadIdx.x >> 5;
  const int lane = threadIdx.x & 31;
  const int nlo = lane & 15, hi = lane >> 4;
  const int qs = (blockIdx.y * 8 + wave) * 16;

  __shared__ unsigned short Ps[8][16][40];   // per-wave 16x32 P tile (+pad)

  const unsigned short* kbase = k + ((size_t)(b * TT)) * EE + h * DD;
  const unsigned short* vbase = vt + ((size_t)((b * HH + h) * DD)) * TT;

  Frag16 qf[2];
  {
    const int m = qs + nlo;
    const unsigned short* qrow = q + ((size_t)(b * TT + m)) * EE + h * DD;
    const int kb = hi * 8;
#pragma unroll
    for (int c = 0; c < 2; ++c) {
      qf[c].q[0] = *(const uint4*)(qrow + c * 32 + kb);
      qf[c].q[1] = *(const uint4*)(qrow + c * 32 + 16 + kb);
    }
  }
  Frag16 onesf;   // all-ones bf16 B fragment (1.0 = 0x3F80)
  onesf.q[0] = make_uint4(0x3F803F80u, 0x3F803F80u, 0x3F803F80u, 0x3F803F80u);
  onesf.q[1] = onesf.q[0];

  v8f o[4];
#pragma unroll
  for (int nt = 0; nt < 4; ++nt)
    o[nt] = (v8f){0.f, 0.f, 0.f, 0.f, 0.f, 0.f, 0.f, 0.f};
  float mrow[8], lrow[8];
#pragma unroll
  for (int r = 0; r < 8; ++r) { mrow[r] = -1e30f; lrow[r] = 0.f; }

  const int nfull = qs >> 5;                 // interior (never-masked) blocks
  for (int jb = 0; jb < nfull; ++jb)
    attn_step<false>(jb * 32, qs, nlo, hi, kbase, vbase, qf, onesf,
                     o, mrow, lrow, Ps[wave]);
  attn_step<true>(nfull * 32, qs, nlo, hi, kbase, vbase, qf, onesf,
                  o, mrow, lrow, Ps[wave]);  // the one diagonal block

  // Normalize and write y[b, qs+row, h*D + col]
#pragma unroll
  for (int r = 0; r < 8; ++r) {
    const float inv = 1.f / lrow[r];
    const size_t base = ((size_t)(b * TT + qs + r + 8 * hi)) * EE + h * DD;
#pragma unroll
    for (int nt = 0; nt < 4; ++nt)
      y[base + nt * 16 + nlo] = o[nt][r] * inv;
  }
}

// ---------------------------------------------------------------------------
extern "C" void kernel_launch(void* const* d_in, const int* in_sizes, int n_in,
                              void* d_out, int out_size, void* d_ws, size_t ws_size,
                              hipStream_t stream) {
  (void)in_sizes; (void)n_in; (void)out_size; (void)ws_size;

  const int*   tokens  = (const int*)  d_in[0];
  const float* tok_emb = (const float*)d_in[1];
  const float* pos_emb = (const float*)d_in[2];
  const float* ln1_g   = (const float*)d_in[3];
  const float* ln1_b   = (const float*)d_in[4];
  const float* Wq      = (const float*)d_in[5];
  const float* bq      = (const float*)d_in[6];
  const float* Wk      = (const float*)d_in[7];
  const float* bk      = (const float*)d_in[8];
  const float* Wv      = (const float*)d_in[9];
  const float* bv      = (const float*)d_in[10];
  const float* Wo      = (const float*)d_in[11];
  const float* bo      = (const float*)d_in[12];
  const float* ln2_g   = (const float*)d_in[13];
  const float* ln2_b   = (const float*)d_in[14];
  const float* Wfc     = (const float*)d_in[15];
  const float* bfc     = (const float*)d_in[16];
  const float* Wpr     = (const float*)d_in[17];
  const float* bpr     = (const float*)d_in[18];
  const float* lnf_g   = (const float*)d_in[19];
  const float* lnf_b   = (const float*)d_in[20];
  const float* Whead   = (const float*)d_in[21];

  const size_t NTOK = (size_t)BB * TT;       // 4096 token rows
  float* ws = (float*)d_ws;
  float* x  = ws;
  float* ln = x  + NTOK * EE;
  float* qb = ln + NTOK * EE;
  float* kb = qb + NTOK * EE;
  float* vb = kb + NTOK * EE;
  float* yb = vb + NTOK * EE;
  float* hb = yb + NTOK * EE;                // NTOK * FFN floats
  unsigned short* qbf = (unsigned short*)(hb + NTOK * FFN);
  unsigned short* kbf = qbf + NTOK * EE;
  unsigned short* vtb = kbf + NTOK * EE;     // [B,H,D,T] bf16

  embed_kernel<<<dim3((unsigned)NTOK), 256, 0, stream>>>(tokens, tok_emb, pos_emb, x);

  const dim3 gE(EE / 128,  (unsigned)(NTOK / 128));   // (8, 32)
  const dim3 gF(FFN / 128, (unsigned)(NTOK / 128));   // (32, 32)
  const dim3 gA(BB * HH, TT / 128);                   // (64, 8)
  const unsigned gC = (unsigned)(NTOK * EE / (8 * 256));  // convert blocks
  const dim3 gV(BB * HH, TT / 64);                    // (64, 16)

  for (int l = 0; l < LLY; ++l) {
    layernorm_kernel<<<(unsigned)NTOK, 256, 0, stream>>>(x, ln1_g + l * EE, ln1_b + l * EE, ln);
    gemm_kernel<0, true, false><<<gE, 256, 0, stream>>>(
        ln, Wq + (size_t)l * EE * EE, bq + l * EE, nullptr, qb, (int)NTOK, EE, EE);
    gemm_kernel<0, true, false><<<gE, 256, 0, stream>>>(
        ln, Wk + (size_t)l * EE * EE, bk + l * EE, nullptr, kb, (int)NTOK, EE, EE);
    gemm_kernel<0, true, false><<<gE, 256, 0, stream>>>(
        ln, Wv + (size_t)l * EE * EE, bv + l * EE, nullptr, vb, (int)NTOK, EE, EE);
    convert_kernel<true ><<<gC, 256, 0, stream>>>(qb, qbf);   // folds 1/sqrt(D)
    convert_kernel<false><<<gC, 256, 0, stream>>>(kb, kbf);
    v_transpose_kernel<<<gV, 256, 0, stream>>>(vb, vtb);
    attn_kernel<<<gA, 256, 0, stream>>>(qbf, kbf, vtb, yb);
    gemm_kernel<0, true, true><<<gE, 256, 0, stream>>>(
        yb, Wo + (size_t)l * EE * EE, bo + l * EE, x, x, (int)NTOK, EE, EE);
    layernorm_kernel<<<(unsigned)NTOK, 256, 0, stream>>>(x, ln2_g + l * EE, ln2_b + l * EE, ln);
    gemm_kernel<1, true, false><<<gF, 256, 0, stream>>>(
        ln, Wfc + (size_t)l * EE * FFN, bfc + l * FFN, nullptr, hb, (int)NTOK, FFN, EE);
    gemm_kernel<0, true, true><<<gE, 256, 0, stream>>>(
        hb, Wpr + (size_t)l * FFN * EE, bpr + l * EE, x, x, (int)NTOK, EE, FFN);
  }
  layernorm_kernel<<<(unsigned)NTOK, 256, 0, stream>>>(x, lnf_g, lnf_b, ln);
  gemm_kernel<0, false, false><<<gE, 256, 0, stream>>>(
      ln, Whead, nullptr, nullptr, (float*)d_out, (int)NTOK, EE, EE);
}
